// QuantizationLayer_44117904064642
// MI455X (gfx1250) — matrix operand order repaired and checked by reading, not compiled, not run on hardware
//
#include <hip/hip_runtime.h>
#include <stdint.h>

// ---------------- problem constants (match reference) ----------------
#define HH 256
#define WW 256
#define BB 32
#define SS 16
#define RAND_START 6
#define FRAGN 4

// ---------------- workspace layout (bytes) ----------------
// bitgrid: (S,B,H,W) bit-packed along x : 16*32*256*8 u32 = 4,194,304 B
// erb16  : (15,B,H, W/16) u16 masks     : 15*32*256*16 u16 = 3,932,160 B
// tilesum: per-(s,b,tile) erosion sums  : 480*256 f32      =   491,520 B
// part   : per-redblock batch stats     : 2048*32*3 f32    =   786,432 B
// tmax   : 1 u32 (+pad)
// params : 32*3 i32 (smax, dx, dy)
#define OFF_BITGRID  0u
#define OFF_ERB      4194304u
#define OFF_TILESUM  8126464u
#define OFF_PART     8617984u
#define OFF_TMAX     9404416u
#define OFF_PARAMS   9404672u
#define WS_TOTAL     9405184u
#define WS_WORDS     (WS_TOTAL / 4u)
#define NBLK_RED     2048

typedef __attribute__((ext_vector_type(16))) _Float16 v16h;
typedef __attribute__((ext_vector_type(8)))  float    v8f;

// ---------------- K0: zero workspace ----------------
__global__ __launch_bounds__(256) void k0_zero(uint32_t* __restrict__ ws) {
  unsigned i = blockIdx.x * 256u + threadIdx.x;
  if (i < WS_WORDS) ws[i] = 0u;
}

// ---------------- K1: t-max + per-batch x/y/count partials ----------------
// Events are sorted by batch -> waves are almost always batch-uniform:
// aggregate with wave32 shfl_xor trees and do 3 shared atomics per WAVE.
__global__ __launch_bounds__(256) void k1_reduce(const float* __restrict__ ev, int N,
                                                 float* __restrict__ part,
                                                 int* __restrict__ tmax) {
  __shared__ float sx[BB], sy[BB], sc[BB];
  __shared__ int smax;
  const int t = threadIdx.x;
  const int lane = t & 31;
  if (t < BB) { sx[t] = 0.f; sy[t] = 0.f; sc[t] = 0.f; }
  if (t == 0) smax = 0;
  __syncthreads();
  int lmax = 0;
  for (int i0 = blockIdx.x * 256; i0 < N; i0 += NBLK_RED * 256) {  // block-uniform trip count
    int i = i0 + t;
    bool valid = i < N;
    if (!__any(valid)) continue;           // lanes hold consecutive i -> wave-uniform skip
    float x = 0.f, y = 0.f, cw = 0.f;
    int bi = 0;
    if (valid) {
      x = ev[i * 5 + 0];
      y = ev[i * 5 + 1];
      int ti = __float_as_int(ev[i * 5 + 2]);   // t > 0 -> int bits order-preserving
      lmax = ti > lmax ? ti : lmax;
      bi = (int)ev[i * 5 + 4];
      cw = 1.f;
    }
    int b0 = __shfl(bi, 0, 32);            // lane 0 is valid whenever any lane is
    bool uni = __all(!valid || bi == b0);
    if (uni) {
      for (int off = 16; off; off >>= 1) {
        x  += __shfl_xor(x, off, 32);
        y  += __shfl_xor(y, off, 32);
        cw += __shfl_xor(cw, off, 32);
      }
      if (lane == 0) {
        atomicAdd(&sx[b0], x);
        atomicAdd(&sy[b0], y);
        atomicAdd(&sc[b0], cw);
      }
    } else if (valid) {                    // rare: batch boundary inside the wave
      atomicAdd(&sx[bi], x);
      atomicAdd(&sy[bi], y);
      atomicAdd(&sc[bi], 1.f);
    }
  }
  atomicMax(&smax, lmax);
  __syncthreads();
  if (t < BB) {
    int o = (blockIdx.x * BB + t) * 3;
    part[o + 0] = sx[t];
    part[o + 1] = sy[t];
    part[o + 2] = sc[t];
  }
  if (t == 0) atomicMax(tmax, smax);
}

// ---------------- K2: boolean scatter into bit-packed voxel grid ----------------
__global__ __launch_bounds__(256) void k2_scatter(const float* __restrict__ ev, int N,
                                                  const int* __restrict__ tmax_i,
                                                  uint32_t* __restrict__ bitgrid) {
  int i = blockIdx.x * 256 + threadIdx.x;
  if (i >= N) return;
  float tmax = __int_as_float(*tmax_i) * 1.001f;
  float x = ev[i * 5 + 0];
  float y = ev[i * 5 + 1];
  float tt = ev[i * 5 + 2];
  int   bi = (int)ev[i * 5 + 4];
  int ts = (int)((tt / tmax) * (float)SS);
  ts = ts < 0 ? 0 : (ts > SS - 1 ? SS - 1 : ts);
  int xi = (int)x, yi = (int)y;
  uint32_t wi = ((uint32_t)(ts * BB + bi) * 256u + (uint32_t)yi) * 8u + (uint32_t)(xi >> 5);
  atomicOr(&bitgrid[wi], 1u << (xi & 31));
}

// 18-bit window for gx = x0-1 .. x0+16 in bits 0..17 (zero outside [0,256))
// g points to one 8-word (256-bit) image row of one slice.
__device__ __forceinline__ uint32_t window18(const uint32_t* __restrict__ g, int x0) {
  if (x0 == 0) return g[0] << 1;           // bit 0 (gx=-1) = 0
  int lo = x0 - 1;
  int w = lo >> 5, sh = lo & 31;
  uint64_t pair = (uint64_t)g[w];
  if (w + 1 < 8) pair |= ((uint64_t)g[w + 1]) << 32;  // x0=240: beyond-image bits read 0
  return (uint32_t)(pair >> sh);
}

// ---------------- K3: separable 3x3 erosion; row pass via WMMA f16 ----------------
// One wave per 16x16 output tile. Tiles: (s in 0..14, b in 0..31, 16x16 tile grid).
// a = dil[s] + 8*dil[s+1] in {-9,-7,7,9};  er = relu((sum3x3(a) - 2*center)/16).
// All conv operands are small integers -> exact in f16 with f32 accumulation.
__global__ __launch_bounds__(256) void k3_conv(const uint32_t* __restrict__ bitgrid,
                                               uint16_t* __restrict__ erb,
                                               float* __restrict__ tilesum) {
  __shared__ float a_lds[8][18][20];   // 18x18 halo tile of 'a' (cols 0..17)
  __shared__ float c_lds[8][16][20];   // column sums, cols 0..17 valid
  const int lane = threadIdx.x & 31;
  const int w    = threadIdx.x >> 5;
  const int T    = blockIdx.x * 8 + w;           // 0 .. 122879
  const int tile = T & 255;
  const int tx = tile & 15, ty = tile >> 4;
  const int sb = T >> 8;                         // 0..479
  const int s = sb >> 5, b = sb & 31;
  const int x0 = tx * 16, y0 = ty * 16;

  // phase 1: lanes 0..17 build one halo row each of 'a' (zero outside image)
  if (lane < 18) {
    int y = y0 - 1 + lane;
    if ((unsigned)y < 256u) {
      const uint32_t* g0 = bitgrid + ((uint32_t)(s * BB + b) * 256u + (uint32_t)y) * 8u;
      const uint32_t* g1 = bitgrid + ((uint32_t)((s + 1) * BB + b) * 256u + (uint32_t)y) * 8u;
      uint32_t r0 = window18(g0, x0);
      uint32_t r1 = window18(g1, x0);
      for (int j = 0; j < 18; ++j) {
        int gx = x0 - 1 + j;
        float av = 0.f;
        if ((unsigned)gx < 256u) {
          int b0 = (int)((r0 >> j) & 1u);
          int b1 = (int)((r1 >> j) & 1u);
          av = (float)(2 * b0 + 16 * b1 - 9);
        }
        a_lds[w][lane][j] = av;
      }
    } else {
      for (int j = 0; j < 18; ++j) a_lds[w][lane][j] = 0.f;
    }
  }
  __syncthreads();

  // phase 2: column pass C[i][j] = a[i][j]+a[i+1][j]+a[i+2][j]  (16x18, 9 per lane)
  for (int q = 0; q < 9; ++q) {
    int idx = lane * 9 + q;          // 0..287
    int i = idx / 18, j = idx - i * 18;
    c_lds[w][i][j] = a_lds[w][i][j] + a_lds[w][i + 1][j] + a_lds[w][i + 2][j];
  }
  __syncthreads();

  // phase 3: row pass as WMMA:  Z(16x16) = C(16x32, zero-padded) x Band(32x16)
  // A layout (16-bit A 16x32): lane m=lane&15; VGPR r<4 -> K=2r(+1), r>=4 -> K=16+2(r-4)(+1); +8 for lanes>=16
  // B layout (32x16 f16): same K order as A (VGPR r<4 -> K=2r(+1), r>=4 -> K=16+2(r-4)(+1); +8 for lanes>=16); N=lane&15
  const int m = lane & 15;
  const int hiA = (lane >> 4) ? 8 : 0;
  const int hiB = (lane >> 4) ? 16 : 0;
  v16h Af, Bf;
#pragma unroll
  for (int e = 0; e < 16; ++e) {
    int r = e >> 1, h = e & 1;
    int ka = ((r < 4) ? 2 * r : 16 + 2 * (r - 4)) + hiA + h;
    Af[e] = (ka < 18) ? (_Float16)c_lds[w][m][ka] : (_Float16)0.0f;
    int kb = ((e < 8) ? e : 8 + e) + hiA;
    Bf[e] = (kb < 18 && kb >= m && kb <= m + 2) ? (_Float16)1.0f : (_Float16)0.0f;
  }
  v8f acc = {0.f, 0.f, 0.f, 0.f, 0.f, 0.f, 0.f, 0.f};
  acc = __builtin_amdgcn_wmma_f32_16x16x32_f16(false, Af, false, Bf, (short)0, acc,
                                               false, false);

  // phase 4: er = relu((Z - 2*center)/16); D layout: n=lane&15, m=(lane>=16?8:0)+v
  const int mrow0 = (lane >> 4) ? 8 : 0;
  float er[8];
  float psum = 0.f;
#pragma unroll
  for (int v = 0; v < 8; ++v) {
    float center = a_lds[w][mrow0 + v + 1][m + 1];
    float z = acc[v] - 2.f * center;
    float e = z > 0.f ? z * 0.0625f : 0.f;
    er[v] = e;
    psum += e;
  }
  for (int off = 16; off; off >>= 1) psum += __shfl_xor(psum, off, 32);
  if (lane == 0) tilesum[T] = psum;   // T == sb*256 + tile

  // eroded boolean planes: ballot gives row v (bits 0..15) and row v+8 (bits 16..31)
  const uint32_t ebase = (uint32_t)(s * BB + b) * 256u * 16u;
  const uint32_t xt = (uint32_t)(x0 >> 4);
#pragma unroll
  for (int v = 0; v < 8; ++v) {
    unsigned mask = (unsigned)__ballot(er[v] > 0.f);
    if (lane == 0) {
      erb[ebase + (uint32_t)(y0 + v) * 16u + xt]     = (uint16_t)(mask & 0xFFFFu);
      erb[ebase + (uint32_t)(y0 + 8 + v) * 16u + xt] = (uint16_t)(mask >> 16);
    }
  }
}

// ---------------- K4: argmax over slice sums + per-batch shift params ----------------
__global__ __launch_bounds__(512) void k4_final(const float* __restrict__ tilesum,
                                                const float* __restrict__ part,
                                                int* __restrict__ params) {
  __shared__ float ssum[480];
  int t = threadIdx.x;
  if (t < 480) {
    float acc = 0.f;
    const float* p = tilesum + t * 256;
    for (int i = 0; i < 256; ++i) acc += p[i];
    ssum[t] = acc;
  }
  __syncthreads();
  if (t < BB) {
    int best = 0;
    float bv = ssum[t];
    for (int s = 1; s < SS - 1; ++s) {
      float v = ssum[s * BB + t];
      if (v > bv) { bv = v; best = s; }
    }
    float sx = 0.f, sy = 0.f, sc = 0.f;
    for (int blk = 0; blk < NBLK_RED; ++blk) {
      int o = (blk * BB + t) * 3;
      sx += part[o + 0];
      sy += part[o + 1];
      sc += part[o + 2];
    }
    float xm = sx / sc, ym = sy / sc;
    params[t * 3 + 0] = best;
    params[t * 3 + 1] = (int)floorf((float)(WW / 2) - xm);
    params[t * 3 + 2] = (int)floorf((float)(HH / 2) - ym);
  }
}

// ---------------- K5: assemble shifted (B,3,H,W) output ----------------
__global__ __launch_bounds__(256) void k5_out(const uint32_t* __restrict__ bitgrid,
                                              const uint16_t* __restrict__ erb,
                                              const int* __restrict__ params,
                                              float* __restrict__ out) {
  int idx = blockIdx.x * 256 + threadIdx.x;   // < 32*256*256
  int b = idx >> 16;
  int y = (idx >> 8) & 255;
  int x = idx & 255;
  int smax = params[b * 3 + 0];
  int dx   = params[b * 3 + 1];
  int dy   = params[b * 3 + 2];
  int ys = y - dy, xs = x - dx;
  float c0 = 0.f, c1 = 0.f, c2 = 0.f;
  if ((unsigned)ys < 256u && (unsigned)xs < 256u) {
    // c1 (slices 6..9) and c2 (slices 0..14) from eroded bit planes
    int wq = xs >> 4, bq = xs & 15;
    for (int s = 0; s < SS - 1; ++s) {
      int bit = (int)((erb[((uint32_t)(s * BB + b) * 256u + (uint32_t)ys) * 16u + wq] >> bq) & 1u);
      c2 += (float)bit;
      if (s >= RAND_START && s < RAND_START + FRAGN) c1 += (float)bit;
    }
    // c0: recompute erosion at the argmax slice from voxel bits
    const uint32_t* g0 = bitgrid + (uint32_t)(smax * BB + b) * 256u * 8u;
    const uint32_t* g1 = bitgrid + (uint32_t)((smax + 1) * BB + b) * 256u * 8u;
    float sum = 0.f, center = 0.f;
    for (int d0 = -1; d0 <= 1; ++d0) {
      int yy = ys + d0;
      if ((unsigned)yy >= 256u) continue;
      for (int d1 = -1; d1 <= 1; ++d1) {
        int xx = xs + d1;
        if ((unsigned)xx >= 256u) continue;
        int wi = yy * 8 + (xx >> 5), bt = xx & 31;
        int b0 = (int)((g0[wi] >> bt) & 1u);
        int b1 = (int)((g1[wi] >> bt) & 1u);
        float a = (float)(2 * b0 + 16 * b1 - 9);
        sum += a;
        if (d0 == 0 && d1 == 0) center = a;
      }
    }
    float z = sum - 2.f * center;
    c0 = z > 0.f ? z * 0.0625f : 0.f;
  }
  int ob = ((b * 3) * 256 + y) * 256 + x;
  out[ob]           = c0;
  out[ob + 65536]   = c1;
  out[ob + 131072]  = c2;
}

// ---------------- host launcher ----------------
extern "C" void kernel_launch(void* const* d_in, const int* in_sizes, int n_in,
                              void* d_out, int out_size, void* d_ws, size_t ws_size,
                              hipStream_t stream) {
  (void)n_in; (void)out_size; (void)ws_size;
  const float* events = (const float*)d_in[0];
  const int N = in_sizes[0] / 5;               // 2,000,000 events
  float* out = (float*)d_out;

  uint8_t* ws = (uint8_t*)d_ws;
  uint32_t* bitgrid = (uint32_t*)(ws + OFF_BITGRID);
  uint16_t* erb     = (uint16_t*)(ws + OFF_ERB);
  float*    tilesum = (float*)   (ws + OFF_TILESUM);
  float*    part    = (float*)   (ws + OFF_PART);
  int*      tmax    = (int*)     (ws + OFF_TMAX);
  int*      params  = (int*)     (ws + OFF_PARAMS);

  // K0: zero workspace
  k0_zero<<<(WS_WORDS + 255u) / 256u, 256, 0, stream>>>((uint32_t*)ws);
  // K1: reductions (wave-aggregated)
  k1_reduce<<<NBLK_RED, 256, 0, stream>>>(events, N, part, tmax);
  // K2: bit scatter
  k2_scatter<<<(N + 255) / 256, 256, 0, stream>>>(events, N, tmax, bitgrid);
  // K3: erosion stencil (WMMA row pass): 15*32*256 tiles, 8 waves/block
  k3_conv<<<(15 * BB * 256) / 8, 256, 0, stream>>>(bitgrid, erb, tilesum);
  // K4: argmax + shift params (single block, fixed-order sums -> deterministic)
  k4_final<<<1, 512, 0, stream>>>(tilesum, part, params);
  // K5: shifted output
  k5_out<<<(BB * HH * WW) / 256, 256, 0, stream>>>(bitgrid, erb, params, out);
}
